// TokenMixingMoE_5652176961934
// MI455X (gfx1250) — matrix operand order repaired and checked by reading, not compiled, not run on hardware
//
#include <hip/hip_runtime.h>
#include <hip/hip_bf16.h>

// ---- problem constants (match reference) ----
#define E_   8
#define N_   8192
#define D_   1024
#define I_   4096
#define EPS_ 1e-5f

typedef __attribute__((ext_vector_type(16))) __bf16 v16bf;
typedef __attribute__((ext_vector_type(8)))  float  v8f;
typedef __attribute__((ext_vector_type(4)))  float  fvec4;
typedef __attribute__((ext_vector_type(2)))  __bf16 bf16x2;
typedef __attribute__((ext_vector_type(4)))  __bf16 bf16x4;
typedef __attribute__((ext_vector_type(8)))  __bf16 bf16x8;

__device__ __forceinline__ float gelu_f(float v) {
    return 0.5f * v * (1.0f + erff(v * 0.70710678118654752f));
}

// Async global->LDS 16B copy (CDNA5, ASYNCcnt-tracked).
__device__ __forceinline__ void async_copy_b128(const __bf16* ga, void* lds_ptr) {
    unsigned lds_off = (unsigned)(unsigned long long)lds_ptr;
    asm volatile("global_load_async_to_lds_b128 %0, %1, off"
                 :: "v"(lds_off), "v"(ga) : "memory");
}
__device__ __forceinline__ void wait_async0() {
    asm volatile("s_wait_asynccnt 0x0" ::: "memory");
}

// -------------------------------------------------------------------------
// Stage 1 (fused): gating + LN1 stats + A1 = gelu(LN1(x)) in bf16 for the
// two selected experts of each token. One wave32 per token; x row stays in
// registers (read once). A1 row id = (token<<1)|k -- deterministic.
// -------------------------------------------------------------------------
__global__ __launch_bounds__(256) void gate_ln1_a1_kernel(
    const float* __restrict__ x,  const float* __restrict__ gw,
    const float* __restrict__ gb, const float* __restrict__ g1,
    const float* __restrict__ b1,
    int* __restrict__ counts, int* __restrict__ entries,
    float* __restrict__ gatev, int* __restrict__ eid,
    __bf16* __restrict__ A1)
{
    const int wave = threadIdx.x >> 5, lane = threadIdx.x & 31;
    const int n = blockIdx.x * 8 + wave;
    const fvec4* xp = (const fvec4*)(x + (size_t)n * D_);

    fvec4 xv[8];
    float s = 0.f, ss = 0.f;
    float dot[E_];
#pragma unroll
    for (int e = 0; e < E_; ++e) dot[e] = 0.f;

#pragma unroll
    for (int j = 0; j < 8; ++j) {
        xv[j] = xp[lane + j * 32];
#pragma unroll
        for (int c = 0; c < 4; ++c) { s += xv[j][c]; ss += xv[j][c] * xv[j][c]; }
#pragma unroll
        for (int e = 0; e < E_; ++e) {
            fvec4 gv = *((const fvec4*)(gw + (size_t)e * D_) + (lane + j * 32));
#pragma unroll
            for (int c = 0; c < 4; ++c) dot[e] += xv[j][c] * gv[c];
        }
    }
#pragma unroll
    for (int off = 16; off > 0; off >>= 1) {
        s  += __shfl_xor(s,  off, 32);
        ss += __shfl_xor(ss, off, 32);
#pragma unroll
        for (int e = 0; e < E_; ++e) dot[e] += __shfl_xor(dot[e], off, 32);
    }
    const float m  = s * (1.0f / D_);
    const float rs = rsqrtf(ss * (1.0f / D_) - m * m + EPS_);
    float lg[E_];
#pragma unroll
    for (int e = 0; e < E_; ++e) lg[e] = dot[e] + gb[e];
    int i0 = 0;
#pragma unroll
    for (int e = 1; e < E_; ++e) if (lg[e] > lg[i0]) i0 = e;
    int i1 = (i0 == 0) ? 1 : 0;
#pragma unroll
    for (int e = 0; e < E_; ++e) if (e != i0 && lg[e] > lg[i1]) i1 = e;

    if (lane == 0) {
        gatev[2 * n + 0] = lg[i0];
        gatev[2 * n + 1] = lg[i1];
        eid[2 * n + 0] = i0;
        eid[2 * n + 1] = i1;
        int p0 = atomicAdd(&counts[i0], 1);
        entries[i0 * N_ + p0] = (n << 1);
        int p1 = atomicAdd(&counts[i1], 1);
        entries[i1 * N_ + p1] = (n << 1) | 1;
    }

#pragma unroll
    for (int k01 = 0; k01 < 2; ++k01) {
        const int ex = k01 ? i1 : i0;
        const float* gp = g1 + (size_t)ex * D_;
        const float* bp = b1 + (size_t)ex * D_;
        __bf16* dst = A1 + ((size_t)(2 * n + k01)) * D_;
#pragma unroll
        for (int j = 0; j < 8; ++j) {
            int i4 = lane + j * 32;
            fvec4 gv = *(const fvec4*)(gp + i4 * 4);
            fvec4 bv = *(const fvec4*)(bp + i4 * 4);
            bf16x4 o;
#pragma unroll
            for (int c = 0; c < 4; ++c)
                o[c] = (__bf16)gelu_f((xv[j][c] - m) * rs * gv[c] + bv[c]);
            *(bf16x4*)(dst + i4 * 4) = o;
        }
    }
}

// ---- software-pipelined tile fill --------------------------------------
// A: 128x32 bf16 copy via async-to-LDS (2 x 16B / thread, no VGPR stage).
// B: 128x32 bf16 from fp32 (issue loads before WMMA, convert+store after).
#define A_ASYNC(buf, kk)                                                     \
    do {                                                                     \
        async_copy_b128(asrc0 + (kk), &As[buf][arowA][ao8]);                 \
        async_copy_b128(asrc1 + (kk), &As[buf][arowB][ao8]);                 \
    } while (0)

#define B_ISSUE(kk, KDIM)                                                    \
    do {                                                                     \
        nb0 = *(const fvec4*)(bsrc0 + (kk));                                 \
        nb1 = *(const fvec4*)(bsrc0 + (kk) + 4);                             \
        nb2 = *(const fvec4*)(bsrc1 + (kk));                                 \
        nb3 = *(const fvec4*)(bsrc1 + (kk) + 4);                             \
        if ((kk) + 32 < (KDIM)) {                                            \
            __builtin_prefetch(bsrc0 + (kk) + 32, 0, 1);                     \
            __builtin_prefetch(bsrc1 + (kk) + 32, 0, 1);                     \
        }                                                                    \
    } while (0)

#define B_COMMIT(buf)                                                        \
    do {                                                                     \
        bf16x8 bv_, cv_;                                                     \
        _Pragma("unroll")                                                    \
        for (int c_ = 0; c_ < 4; ++c_) {                                     \
            bv_[c_] = (__bf16)nb0[c_]; bv_[4 + c_] = (__bf16)nb1[c_];        \
            cv_[c_] = (__bf16)nb2[c_]; cv_[4 + c_] = (__bf16)nb3[c_];        \
        }                                                                    \
        *(bf16x8*)&Bs[buf][brow][ao8] = bv_;                                 \
        *(bf16x8*)&Bs[buf][brow + 64][ao8] = cv_;                            \
    } while (0)

// ---- fragment load + 2x4 WMMA (wave tile 32Mx64N) ----------------------
#define WMMA_STEP(buf)                                                       \
    do {                                                                     \
        v16bf af_[2], bf_[4];                                                \
        _Pragma("unroll")                                                    \
        for (int sm_ = 0; sm_ < 2; ++sm_) {                                  \
            int row_ = waveM * 32 + sm_ * 16 + r;                            \
            _Pragma("unroll")                                                \
            for (int v_ = 0; v_ < 8; ++v_) {                                 \
                int kb_ = ((v_ & 3) << 1) + ((v_ >> 2) << 4) + (hi << 3);    \
                bf16x2 p_ = *(const bf16x2*)&As[buf][row_][kb_];             \
                af_[sm_][2 * v_] = p_.x; af_[sm_][2 * v_ + 1] = p_.y;        \
            }                                                                \
        }                                                                    \
        _Pragma("unroll")                                                    \
        for (int sn_ = 0; sn_ < 4; ++sn_) {                                  \
            int col_ = waveN * 64 + sn_ * 16 + r;                            \
            _Pragma("unroll")                                                \
            for (int v_ = 0; v_ < 8; ++v_) {                                 \
                int kk_ = ((v_ & 3) << 1) + ((v_ >> 2) << 4) + (hi << 3);    \
                bf16x2 p_ = *(const bf16x2*)&Bs[buf][col_][kk_];             \
                bf_[sn_][2 * v_] = p_.x; bf_[sn_][2 * v_ + 1] = p_.y;        \
            }                                                                \
        }                                                                    \
        _Pragma("unroll")                                                    \
        for (int sm_ = 0; sm_ < 2; ++sm_)                                    \
            _Pragma("unroll")                                                \
            for (int sn_ = 0; sn_ < 4; ++sn_)                                \
                acc[sm_][sn_] = __builtin_amdgcn_wmma_f32_16x16x32_bf16(     \
                    false, af_[sm_], false, bf_[sn_], (short)0,              \
                    acc[sm_][sn_], false, false);                            \
    } while (0)

// -------------------------------------------------------------------------
// Grouped GEMM 1: H[row,:] = A1[row,:] @ W1[e]^T  (bf16 WMMA, fp32 acc)
// 128(M) x 128(N) block tile, 8 waves as 4(M)x2(N), double-buffered LDS,
// B loads software-pipelined behind the WMMAs.
// -------------------------------------------------------------------------
__global__ __launch_bounds__(256) void gemm1_kernel(
    const __bf16* __restrict__ A1, const float* __restrict__ w1,
    const int* __restrict__ counts, const int* __restrict__ entries,
    __bf16* __restrict__ H)
{
    const int e   = blockIdx.z;
    const int cnt = counts[e];
    const int m0  = blockIdx.x * 128;
    if (m0 >= cnt) return;
    const int n0  = blockIdx.y * 128;

    __shared__ __bf16 As[2][128][40];
    __shared__ __bf16 Bs[2][128][40];
    __shared__ int    rid_s[128];

    const int tid = threadIdx.x;
    if (tid < 128) {
        int mm = m0 + tid;
        rid_s[tid] = (mm < cnt) ? entries[e * N_ + mm] : -1;
    }
    __syncthreads();

    const int arowA = tid >> 2, arowB = arowA + 64, ao8 = (tid & 3) * 8;
    const int brow  = tid >> 2;
    int aentA = rid_s[arowA]; if (aentA < 0) aentA = 0;
    int aentB = rid_s[arowB]; if (aentB < 0) aentB = 0;
    const __bf16* asrc0 = A1 + (size_t)aentA * D_ + ao8;
    const __bf16* asrc1 = A1 + (size_t)aentB * D_ + ao8;
    const float*  bsrc0 = w1 + ((size_t)e * I_ + (n0 + brow))      * D_ + ao8;
    const float*  bsrc1 = w1 + ((size_t)e * I_ + (n0 + brow + 64)) * D_ + ao8;

    const int lane = tid & 31, wave = tid >> 5;
    const int waveM = wave >> 1, waveN = wave & 1;
    const int r = lane & 15, hi = lane >> 4;

    v8f acc[2][4] = {};
    fvec4 nb0, nb1, nb2, nb3;

    A_ASYNC(0, 0);
    B_ISSUE(0, D_);
    B_COMMIT(0);
    wait_async0();
    __syncthreads();
    for (int k0 = 0; k0 < D_; k0 += 32) {
        const int  cur = (k0 >> 5) & 1, nxt = cur ^ 1;
        const bool hn  = (k0 + 32) < D_;
        if (hn) { A_ASYNC(nxt, k0 + 32); B_ISSUE(k0 + 32, D_); }
        WMMA_STEP(cur);                    // hides B-load latency
        if (hn) B_COMMIT(nxt);
        wait_async0();
        __syncthreads();
    }

    // epilogue: row lookup hoisted to (sm, j); 4 stores share one base
#pragma unroll
    for (int sm = 0; sm < 2; ++sm)
#pragma unroll
        for (int j = 0; j < 8; ++j) {
            int lr  = waveM * 32 + sm * 16 + (hi << 3) + j;
            int ent = rid_s[lr];
            if (ent >= 0) {
                __bf16* hrow = H + (size_t)ent * I_ + n0 + waveN * 64 + r;
#pragma unroll
                for (int sn = 0; sn < 4; ++sn)
                    hrow[sn * 16] = (__bf16)acc[sm][sn][j];
            }
        }
}

// -------------------------------------------------------------------------
// Stage 3 (fused): LN2 stats + in-place H <- gelu(LN2(H)) in bf16.
// -------------------------------------------------------------------------
__global__ __launch_bounds__(256) void ln2_a2_kernel(
    __bf16* __restrict__ H, const float* __restrict__ g2,
    const float* __restrict__ b2ln, const int* __restrict__ eid)
{
    const int row  = blockIdx.x * 8 + (threadIdx.x >> 5);
    const int lane = threadIdx.x & 31;
    __bf16* hp = H + (size_t)row * I_;

    float s = 0.f, ss = 0.f;
    for (int i = lane * 4; i < I_; i += 128) {
        bf16x4 p = *(const bf16x4*)(hp + i);
#pragma unroll
        for (int c = 0; c < 4; ++c) { float f = (float)p[c]; s += f; ss += f * f; }
    }
#pragma unroll
    for (int off = 16; off > 0; off >>= 1) {
        s += __shfl_xor(s, off, 32); ss += __shfl_xor(ss, off, 32);
    }
    const float m  = s * (1.0f / I_);
    const float rs = rsqrtf(ss * (1.0f / I_) - m * m + EPS_);
    const int   ex = eid[row];
    const float* gp = g2   + (size_t)ex * I_;
    const float* bp = b2ln + (size_t)ex * I_;
    for (int i = lane * 4; i < I_; i += 128) {
        bf16x4 p = *(const bf16x4*)(hp + i);
        fvec4 gv = *(const fvec4*)(gp + i);
        fvec4 bv = *(const fvec4*)(bp + i);
        bf16x4 o;
#pragma unroll
        for (int c = 0; c < 4; ++c)
            o[c] = (__bf16)gelu_f(((float)p[c] - m) * rs * gv[c] + bv[c]);
        *(bf16x4*)(hp + i) = o;
    }
}

// -------------------------------------------------------------------------
// Grouped GEMM 2: out[token] += gate * (A2[row,:] @ W2[e]^T + b2[e])
// Exactly 2 atomic adds per (token,d) from 0 -> order-invariant.
// -------------------------------------------------------------------------
__global__ __launch_bounds__(256) void gemm2_kernel(
    const __bf16* __restrict__ H, const float* __restrict__ w2,
    const float* __restrict__ b2, const int* __restrict__ counts,
    const int* __restrict__ entries, const float* __restrict__ gatev,
    float* __restrict__ out)
{
    const int e   = blockIdx.z;
    const int cnt = counts[e];
    const int m0  = blockIdx.x * 128;
    if (m0 >= cnt) return;
    const int d0  = blockIdx.y * 128;

    __shared__ __bf16 As[2][128][40];
    __shared__ __bf16 Bs[2][128][40];
    __shared__ int    rid_s[128];
    __shared__ float  gt_s[128];

    const int tid = threadIdx.x;
    if (tid < 128) {
        int mm = m0 + tid;
        int ent = (mm < cnt) ? entries[e * N_ + mm] : -1;
        rid_s[tid] = ent;
        gt_s[tid]  = (ent >= 0) ? gatev[ent] : 0.f;
    }
    __syncthreads();

    const int arowA = tid >> 2, arowB = arowA + 64, ao8 = (tid & 3) * 8;
    const int brow  = tid >> 2;
    int aentA = rid_s[arowA]; if (aentA < 0) aentA = 0;
    int aentB = rid_s[arowB]; if (aentB < 0) aentB = 0;
    const __bf16* asrc0 = H  + (size_t)aentA * I_ + ao8;
    const __bf16* asrc1 = H  + (size_t)aentB * I_ + ao8;
    const float*  bsrc0 = w2 + ((size_t)e * D_ + (d0 + brow))      * I_ + ao8;
    const float*  bsrc1 = w2 + ((size_t)e * D_ + (d0 + brow + 64)) * I_ + ao8;

    const int lane = tid & 31, wave = tid >> 5;
    const int waveM = wave >> 1, waveN = wave & 1;
    const int r = lane & 15, hi = lane >> 4;

    v8f acc[2][4] = {};
    fvec4 nb0, nb1, nb2, nb3;

    A_ASYNC(0, 0);
    B_ISSUE(0, I_);
    B_COMMIT(0);
    wait_async0();
    __syncthreads();
    for (int k0 = 0; k0 < I_; k0 += 32) {
        const int  cur = (k0 >> 5) & 1, nxt = cur ^ 1;
        const bool hn  = (k0 + 32) < I_;
        if (hn) { A_ASYNC(nxt, k0 + 32); B_ISSUE(k0 + 32, I_); }
        WMMA_STEP(cur);
        if (hn) B_COMMIT(nxt);
        wait_async0();
        __syncthreads();
    }

    // epilogue: b2 preloaded per sn; row lookup hoisted to (sm, j)
    float bb[4];
#pragma unroll
    for (int sn = 0; sn < 4; ++sn)
        bb[sn] = b2[(size_t)e * D_ + d0 + waveN * 64 + sn * 16 + r];
#pragma unroll
    for (int sm = 0; sm < 2; ++sm)
#pragma unroll
        for (int j = 0; j < 8; ++j) {
            int lr  = waveM * 32 + sm * 16 + (hi << 3) + j;
            int ent = rid_s[lr];
            if (ent >= 0) {
                float g = gt_s[lr];
                float* orow = out + (size_t)(ent >> 1) * D_ + d0 + waveN * 64 + r;
#pragma unroll
                for (int sn = 0; sn < 4; ++sn)
                    atomicAdd(orow + sn * 16, g * (acc[sm][sn][j] + bb[sn]));
            }
        }
}

// -------------------------------------------------------------------------
extern "C" void kernel_launch(void* const* d_in, const int* in_sizes, int n_in,
                              void* d_out, int out_size, void* d_ws, size_t ws_size,
                              hipStream_t stream) {
    const float* x    = (const float*)d_in[0];
    const float* gw   = (const float*)d_in[1];
    const float* gb   = (const float*)d_in[2];
    const float* g1   = (const float*)d_in[3];
    const float* b1   = (const float*)d_in[4];
    const float* w1   = (const float*)d_in[5];
    const float* g2   = (const float*)d_in[6];
    const float* b2ln = (const float*)d_in[7];
    const float* w2   = (const float*)d_in[8];
    const float* b2   = (const float*)d_in[9];
    float* out = (float*)d_out;
    (void)in_sizes; (void)n_in; (void)out_size; (void)ws_size;

    char* ws = (char*)d_ws;
    size_t off = 0;
    auto alloc = [&](size_t bytes) {
        void* p = ws + off;
        off = (off + bytes + 255) & ~(size_t)255;
        return p;
    };
    int*    counts  = (int*)   alloc((size_t)E_ * sizeof(int));
    int*    entries = (int*)   alloc((size_t)E_ * N_ * sizeof(int));
    float*  gatev   = (float*) alloc((size_t)2 * N_ * sizeof(float));
    int*    eid     = (int*)   alloc((size_t)2 * N_ * sizeof(int));
    __bf16* A1      = (__bf16*)alloc((size_t)2 * N_ * D_ * sizeof(__bf16));
    __bf16* H       = (__bf16*)alloc((size_t)2 * N_ * I_ * sizeof(__bf16));

    hipMemsetAsync(counts, 0, E_ * sizeof(int), stream);
    hipMemsetAsync(out, 0, (size_t)N_ * D_ * sizeof(float), stream);

    gate_ln1_a1_kernel<<<N_ / 8, 256, 0, stream>>>(x, gw, gb, g1, b1, counts,
                                                   entries, gatev, eid, A1);
    gemm1_kernel<<<dim3(N_ / 128, I_ / 128, E_), 256, 0, stream>>>(
        A1, w1, counts, entries, H);
    ln2_a2_kernel<<<(2 * N_) / 8, 256, 0, stream>>>(H, g2, b2ln, eid);
    gemm2_kernel<<<dim3(N_ / 128, D_ / 128, E_), 256, 0, stream>>>(
        H, w2, b2, counts, entries, gatev, out);
}
